// NAM_52115133170271
// MI455X (gfx1250) — matrix operand order repaired and checked by reading, not compiled, not run on hardware
//
#include <hip/hip_runtime.h>

// ---- problem dims ----
#define B_   4096
#define F_   128
#define U_   512
#define H1_  256
#define H2_  128
#define BM   32          // batch rows per workgroup (2 M-tiles of 16)

// padded LDS row pitches (bf16 elements) to kill bank conflicts on A-frag loads
#define P0   (U_  + 8)   // 520 -> 1040B/row, 16B aligned, bank step 4
#define P1   (H1_ + 8)   // 264 -> 528B/row
#define P2   (H2_ + 8)   // 136

typedef __attribute__((ext_vector_type(16))) __bf16 v16bf;
typedef __attribute__((ext_vector_type(8)))  __bf16 v8bf;
typedef __attribute__((ext_vector_type(8)))  float  v8f;

union AFrag { v16bf v; v8bf h[2]; };

// ---------------------------------------------------------------------------
// Transpose+convert: src [F][R][C] f32  ->  dst [F][C][R] bf16
// 32x32 tiles via LDS so both global read and write are coalesced.
// ---------------------------------------------------------------------------
__global__ __launch_bounds__(256) void nam_transpose_bf16(
    const float* __restrict__ src, __bf16* __restrict__ dst, int R, int C) {
  __shared__ float tile[32][33];
  const int f  = blockIdx.z;
  const int c0 = blockIdx.x * 32;
  const int r0 = blockIdx.y * 32;
  const int tx = threadIdx.x;       // 32
  const int ty = threadIdx.y;       // 8
  const float* s = src + (size_t)f * R * C;
  __bf16*      d = dst + (size_t)f * R * C;
#pragma unroll
  for (int i = 0; i < 4; ++i)
    tile[ty + i * 8][tx] = s[(size_t)(r0 + ty + i * 8) * C + (c0 + tx)];
  __syncthreads();
#pragma unroll
  for (int i = 0; i < 4; ++i)
    d[(size_t)(c0 + ty + i * 8) * R + (r0 + tx)] = (__bf16)tile[tx][ty + i * 8];
}

// ---------------------------------------------------------------------------
// Fused per-feature MLP: one WG = (feature f, 32 batch rows).
// All activations live in LDS as bf16; GEMMs use v_wmma_f32_16x16x32_bf16.
// ---------------------------------------------------------------------------
__global__ __launch_bounds__(256) void nam_mlp_kernel(
    const float*  __restrict__ inputs,
    const float*  __restrict__ w0, const float* __restrict__ b0,
    const __bf16* __restrict__ w1t, const float* __restrict__ b1,
    const __bf16* __restrict__ w2t, const float* __restrict__ b2,
    const float*  __restrict__ w3, const float* __restrict__ b3,
    float* __restrict__ feat_out) {
  __shared__ __bf16 sh0[BM * P0];
  __shared__ __bf16 sh1[BM * P1];
  __shared__ __bf16 sh2[BM * P2];
  __shared__ float  sx[BM];

  const int f     = blockIdx.y;
  const int brow0 = blockIdx.x * BM;
  const int tid   = threadIdx.x;
  const int lane  = tid & 31;
  const int wave  = tid >> 5;       // 0..7

  if (tid < BM) sx[tid] = inputs[(size_t)(brow0 + tid) * F_ + f];
  __syncthreads();

  // ---- layer 0: h0 = relu(x * w0 + b0)  -> LDS [BM x U] bf16 ----
  {
    const float* w0f = w0 + (size_t)f * U_;
    const float* b0f = b0 + (size_t)f * U_;
    for (int idx = tid; idx < BM * U_; idx += 256) {
      const int m = idx >> 9;              // /U_
      const int u = idx & (U_ - 1);
      const float v = fmaf(sx[m], w0f[u], b0f[u]);
      sh0[m * P0 + u] = (__bf16)fmaxf(v, 0.0f);
    }
  }
  __syncthreads();

  // ISA fragment geometry (wave32):
  const int mrow  = lane & 15;         // A row within M-tile
  const int khalf = (lane >> 4) * 8;   // A-frag K sub-offset (lanes 16-31 -> +8)
  const int bk    = (lane >> 4) * 8;   // B-frag K sub-offset (lanes 16-31 -> +8)
  const int ncl   = lane & 15;         // B/C column within N-tile
  const int crow  = (lane >> 4) * 8;   // C-frag M sub-offset

  // ---- layer 1: h1 = relu(h0 @ w1 + b1), 32x256; wave owns 2 N-cols x 2 M-tiles
  {
    v8f acc[2][2] = {};  // [mtile][ncol]
    const __bf16* bp0 = w1t + ((size_t)f * H1_ + (2 * wave + 0) * 16 + ncl) * U_ + bk;
    const __bf16* bp1 = w1t + ((size_t)f * H1_ + (2 * wave + 1) * 16 + ncl) * U_ + bk;
    for (int kb = 0; kb < U_; kb += 32) {
      AFrag a0, a1, bf0, bf1;
      a0.h[0] = *(const v8bf*)&sh0[(0 * 16 + mrow) * P0 + kb + khalf];
      a0.h[1] = *(const v8bf*)&sh0[(0 * 16 + mrow) * P0 + kb + 16 + khalf];
      a1.h[0] = *(const v8bf*)&sh0[(1 * 16 + mrow) * P0 + kb + khalf];
      a1.h[1] = *(const v8bf*)&sh0[(1 * 16 + mrow) * P0 + kb + 16 + khalf];
      bf0.h[0] = *(const v8bf*)(bp0 + kb);
      bf0.h[1] = *(const v8bf*)(bp0 + kb + 16);
      bf1.h[0] = *(const v8bf*)(bp1 + kb);
      bf1.h[1] = *(const v8bf*)(bp1 + kb + 16);
      __builtin_prefetch(bp0 + kb + 32, 0, 1);   // global_prefetch_b8
      acc[0][0] = __builtin_amdgcn_wmma_f32_16x16x32_bf16(false, a0.v, false, bf0.v, (short)0, acc[0][0], false, false);
      acc[1][0] = __builtin_amdgcn_wmma_f32_16x16x32_bf16(false, a1.v, false, bf0.v, (short)0, acc[1][0], false, false);
      acc[0][1] = __builtin_amdgcn_wmma_f32_16x16x32_bf16(false, a0.v, false, bf1.v, (short)0, acc[0][1], false, false);
      acc[1][1] = __builtin_amdgcn_wmma_f32_16x16x32_bf16(false, a1.v, false, bf1.v, (short)0, acc[1][1], false, false);
    }
    const float* b1f = b1 + (size_t)f * H1_;
#pragma unroll
    for (int j = 0; j < 2; ++j) {
      const int   ncolg = (2 * wave + j) * 16 + ncl;
      const float bb    = b1f[ncolg];
#pragma unroll
      for (int mt = 0; mt < 2; ++mt)
#pragma unroll
        for (int r = 0; r < 8; ++r) {
          const float v = acc[mt][j][r] + bb;
          sh1[(mt * 16 + crow + r) * P1 + ncolg] = (__bf16)fmaxf(v, 0.0f);
        }
    }
  }
  __syncthreads();

  // ---- layer 2: h2 = relu(h1 @ w2 + b2), 32x128; wave owns 1 N-col x 2 M-tiles
  {
    v8f acc[2] = {};
    const __bf16* bp = w2t + ((size_t)f * H2_ + wave * 16 + ncl) * H1_ + bk;
    for (int kb = 0; kb < H1_; kb += 32) {
      AFrag a0, a1, bf;
      a0.h[0] = *(const v8bf*)&sh1[(0 * 16 + mrow) * P1 + kb + khalf];
      a0.h[1] = *(const v8bf*)&sh1[(0 * 16 + mrow) * P1 + kb + 16 + khalf];
      a1.h[0] = *(const v8bf*)&sh1[(1 * 16 + mrow) * P1 + kb + khalf];
      a1.h[1] = *(const v8bf*)&sh1[(1 * 16 + mrow) * P1 + kb + 16 + khalf];
      bf.h[0] = *(const v8bf*)(bp + kb);
      bf.h[1] = *(const v8bf*)(bp + kb + 16);
      acc[0] = __builtin_amdgcn_wmma_f32_16x16x32_bf16(false, a0.v, false, bf.v, (short)0, acc[0], false, false);
      acc[1] = __builtin_amdgcn_wmma_f32_16x16x32_bf16(false, a1.v, false, bf.v, (short)0, acc[1], false, false);
    }
    const float* b2f   = b2 + (size_t)f * H2_;
    const int    ncolg = wave * 16 + ncl;
    const float  bb    = b2f[ncolg];
#pragma unroll
    for (int mt = 0; mt < 2; ++mt)
#pragma unroll
      for (int r = 0; r < 8; ++r) {
        const float v = acc[mt][r] + bb;
        sh2[(mt * 16 + crow + r) * P2 + ncolg] = (__bf16)fmaxf(v, 0.0f);
      }
  }
  __syncthreads();

  // ---- layer 3: feat = h2 @ w3 + b3  (tiny dot; 8 threads per row) ----
  {
    const float* w3f = w3 + (size_t)f * H2_;
    const int m  = tid >> 3;            // 0..31
    const int k0 = (tid & 7) * 16;
    float p = 0.0f;
#pragma unroll
    for (int i = 0; i < 16; ++i)
      p = fmaf((float)sh2[m * P2 + k0 + i], w3f[k0 + i], p);
    p += __shfl_xor(p, 4, 32);
    p += __shfl_xor(p, 2, 32);
    p += __shfl_xor(p, 1, 32);
    if ((tid & 7) == 0)
      feat_out[(size_t)(brow0 + m) * F_ + f] = p + b3[f];
  }
}

// ---------------------------------------------------------------------------
// out[b] = bias + sum_f feat[b,f]   (one wave per row)
// ---------------------------------------------------------------------------
__global__ __launch_bounds__(256) void nam_rowsum_kernel(
    const float* __restrict__ feat, const float* __restrict__ bias,
    float* __restrict__ out) {
  const int row  = blockIdx.x * 8 + (threadIdx.x >> 5);
  const int lane = threadIdx.x & 31;
  const float* p = feat + (size_t)row * F_;
  float s = p[lane] + p[lane + 32] + p[lane + 64] + p[lane + 96];
#pragma unroll
  for (int off = 16; off; off >>= 1) s += __shfl_xor(s, off, 32);
  if (lane == 0) out[row] = s + bias[0];
}

// ---------------------------------------------------------------------------
extern "C" void kernel_launch(void* const* d_in, const int* in_sizes, int n_in,
                              void* d_out, int out_size, void* d_ws, size_t ws_size,
                              hipStream_t stream) {
  const float* inputs = (const float*)d_in[0];
  const float* w0     = (const float*)d_in[1];
  const float* b0     = (const float*)d_in[2];
  const float* w1     = (const float*)d_in[3];
  const float* b1     = (const float*)d_in[4];
  const float* w2     = (const float*)d_in[5];
  const float* b2     = (const float*)d_in[6];
  const float* w3     = (const float*)d_in[7];
  const float* b3     = (const float*)d_in[8];
  const float* bias   = (const float*)d_in[9];

  float* out  = (float*)d_out;   // [B]
  float* feat = out + B_;        // [B, F] (second tuple output, flat)

  __bf16* w1t = (__bf16*)d_ws;                         // [F][H1][U] bf16, 32 MB
  __bf16* w2t = w1t + (size_t)F_ * U_ * H1_;           // [F][H2][H1] bf16, 8 MB

  // convert + transpose weights (K-contiguous bf16 for WMMA B-fragments)
  nam_transpose_bf16<<<dim3(H1_ / 32, U_ / 32, F_), dim3(32, 8), 0, stream>>>(w1, w1t, U_, H1_);
  nam_transpose_bf16<<<dim3(H2_ / 32, H1_ / 32, F_), dim3(32, 8), 0, stream>>>(w2, w2t, H1_, H2_);

  // fused MLP
  nam_mlp_kernel<<<dim3(B_ / BM, F_), 256, 0, stream>>>(
      inputs, w0, b0, w1t, b1, w2t, b2, w3, b3, feat);

  // final cross-feature reduction
  nam_rowsum_kernel<<<B_ / 8, 256, 0, stream>>>(feat, bias, out);
}
